// FeaStNet_10737418240590
// MI455X (gfx1250) — matrix operand order repaired and checked by reading, not compiled, not run on hardware
//
#include <hip/hip_runtime.h>
#include <hip/hip_bf16.h>

#define NNODES   50000
#define NEDGES   800000
#define NHEADS   4
#define NGRAPHS  8
#define NCLASSES 10

typedef __attribute__((ext_vector_type(16))) _Float16 v16h;
typedef __attribute__((ext_vector_type(8)))  float    v8f;

// ---------------- fc0: h0 = relu(x @ W + b), x:[N,3], W:[3,16] ----------------
__global__ void k_fc0(const float* __restrict__ x, const float* __restrict__ W,
                      const float* __restrict__ b, float* __restrict__ h0, int n) {
  int i = blockIdx.x * blockDim.x + threadIdx.x;
  if (i >= n) return;
  float x0 = x[i * 3 + 0], x1 = x[i * 3 + 1], x2 = x[i * 3 + 2];
#pragma unroll
  for (int c = 0; c < 16; ++c) {
    float v = x0 * W[0 * 16 + c] + x1 * W[1 * 16 + c] + x2 * W[2 * 16 + c] + b[c];
    h0[i * 16 + c] = v > 0.f ? v : 0.f;
  }
}

// ---------------- degree of dst (shared by both conv layers) ----------------
__global__ void k_deg(const int* __restrict__ dst, float* __restrict__ deg, int e) {
  int i = blockIdx.x * blockDim.x + threadIdx.x;
  if (i < e) atomicAdd(&deg[dst[i]], 1.0f);
}

// ---------------- WMMA GEMM: C[N,OUT] = A[N,KD] @ W[KD,OUT] -----------------
// f16 inputs, f32 accumulate via v_wmma_f32_16x16x32_f16.
// W is staged in LDS *transposed* as [col][32] halfs, K zero-padded to 32, so
// each lane's B fragment is two 16-byte LDS reads (2x ds_load_b128)
// with no lane-dependent predication. A rows load as float4 (global_load_b128).
template <int KD, int OUT>
__global__ void k_gemm_wmma(const float* __restrict__ A,
                            const float* __restrict__ W,
                            float* __restrict__ C, int n) {
  static_assert(KD == 16 || KD == 32, "KD must be 16 or 32");
  __shared__ alignas(32) _Float16 lwT[OUT * 32];  // [col][k], k padded to 32
  for (int idx = threadIdx.x; idx < OUT * 32; idx += blockDim.x) {
    int c = idx >> 5, k = idx & 31;
    lwT[idx] = (k < KD) ? (_Float16)W[k * OUT + c] : (_Float16)0.f;
  }
  __syncthreads();

  int wave = blockIdx.x * (blockDim.x >> 5) + (threadIdx.x >> 5);
  int lane = threadIdx.x & 31;
  if (wave >= (n >> 4)) return;  // n multiple of 16 (50000/16 = 3125)

  // A fragment (ISA 7.12.2): lane holds row (lane&15); K groups split by lane-half
  int row = (wave << 4) + (lane & 15);
  int kb  = (lane >> 4) * 8;
  const float* arow = A + (size_t)row * KD + kb;
  v16h a;
  {
    float4 f0 = *(const float4*)(arow);
    float4 f1 = *(const float4*)(arow + 4);
    float lo[8] = {f0.x, f0.y, f0.z, f0.w, f1.x, f1.y, f1.z, f1.w};
#pragma unroll
    for (int i = 0; i < 8; ++i) a[i] = (_Float16)lo[i];
    if (KD == 32) {
      float4 f2 = *(const float4*)(arow + 16);
      float4 f3 = *(const float4*)(arow + 20);
      float hi[8] = {f2.x, f2.y, f2.z, f2.w, f3.x, f3.y, f3.z, f3.w};
#pragma unroll
      for (int i = 0; i < 8; ++i) a[8 + i] = (_Float16)hi[i];
    } else {
#pragma unroll
      for (int i = 0; i < 8; ++i) a[8 + i] = (_Float16)0.f;  // compile-time pad
    }
  }

  int koff  = (lane >> 4) * 8;  // B fragment: lanes 0-15 K=0..7,16..23, lanes 16-31 K=8..15,24..31
  int rbase = (wave << 4) + ((lane >> 4) * 8);
  float* crow = C + (size_t)rbase * OUT + (lane & 15);
#pragma unroll
  for (int tn = 0; tn < OUT / 16; ++tn) {
    int col = tn * 16 + (lane & 15);
    v16h bm;
#pragma unroll
    for (int i = 0; i < 8; ++i) {
      bm[i]     = lwT[(col << 5) + koff + i];
      bm[8 + i] = lwT[(col << 5) + 16 + koff + i];
    }
    v8f acc = {};
    acc = __builtin_amdgcn_wmma_f32_16x16x32_f16(false, a, false, bm,
                                                 (short)0, acc, false, false);
#pragma unroll
    for (int r = 0; r < 8; ++r)
      crow[(size_t)r * OUT + tn * 16] = acc[r];
  }
}

// ---------------- per-edge head softmax: q[E,4] -----------------------------
template <int CIN>
__global__ void k_q(const float* __restrict__ h, const int* __restrict__ src,
                    const int* __restrict__ dst, const float* __restrict__ u,
                    const float* __restrict__ cbias, float* __restrict__ q, int e) {
  int i = blockIdx.x * blockDim.x + threadIdx.x;
  if (i >= e) return;
  int s = src[i], d = dst[i];
  float d0 = cbias[0], d1 = cbias[1], d2 = cbias[2], d3 = cbias[3];
#pragma unroll
  for (int k = 0; k < CIN; ++k) {
    float diff = h[(size_t)s * CIN + k] - h[(size_t)d * CIN + k];
    d0 += diff * u[k * 4 + 0];
    d1 += diff * u[k * 4 + 1];
    d2 += diff * u[k * 4 + 2];
    d3 += diff * u[k * 4 + 3];
  }
  float m  = fmaxf(fmaxf(d0, d1), fmaxf(d2, d3));
  float e0 = __expf(d0 - m), e1 = __expf(d1 - m);
  float e2 = __expf(d2 - m), e3 = __expf(d3 - m);
  float inv = 1.f / (e0 + e1 + e2 + e3);
  q[i * 4 + 0] = e0 * inv;
  q[i * 4 + 1] = e1 * inv;
  q[i * 4 + 2] = e2 * inv;
  q[i * 4 + 3] = e3 * inv;
}

// ---------------- per-edge combine + scatter: one wave per edge -------------
// msg = sum_h q[h] * xw[src, h, :]; atomicAdd into agg[dst, :]
template <int OUT>
__global__ void k_edge(const float* __restrict__ q, const float* __restrict__ xw,
                       const int* __restrict__ src, const int* __restrict__ dst,
                       float* __restrict__ agg, int e) {
  int gid  = blockIdx.x * blockDim.x + threadIdx.x;
  int edge = gid >> 5;
  int lane = gid & 31;
  if (edge >= e) return;
  int s = src[edge], d = dst[edge];
  float q0 = q[edge * 4 + 0], q1 = q[edge * 4 + 1];
  float q2 = q[edge * 4 + 2], q3 = q[edge * 4 + 3];
  const float* xs = xw + (size_t)s * (4 * OUT);
#pragma unroll
  for (int c = lane; c < OUT; c += 32) {
    float m = q0 * xs[0 * OUT + c] + q1 * xs[1 * OUT + c] +
              q2 * xs[2 * OUT + c] + q3 * xs[3 * OUT + c];
    atomicAdd(&agg[(size_t)d * OUT + c], m);
  }
}

// ---------------- mean-normalize + bias + relu (in place) -------------------
template <int OUT>
__global__ void k_norm(float* __restrict__ agg, const float* __restrict__ deg,
                       const float* __restrict__ b, int n) {
  int i = blockIdx.x * blockDim.x + threadIdx.x;
  if (i >= n * OUT) return;
  int node = i / OUT, c = i - node * OUT;
  float v = agg[i] / fmaxf(deg[node], 1.0f) + b[c];
  agg[i] = v > 0.f ? v : 0.f;
}

// ---------------- global mean pool (hierarchical: LDS then global) ----------
__global__ void k_pool(const float* __restrict__ h2, const int* __restrict__ batch,
                       float* __restrict__ pool, float* __restrict__ cnt, int n) {
  __shared__ float ls[NGRAPHS * 64];
  __shared__ float lc[NGRAPHS];
  for (int i = threadIdx.x; i < NGRAPHS * 64; i += blockDim.x) ls[i] = 0.f;
  if (threadIdx.x < NGRAPHS) lc[threadIdx.x] = 0.f;
  __syncthreads();
  int node = blockIdx.x * blockDim.x + threadIdx.x;
  if (node < n) {
    int g = batch[node];
    atomicAdd(&lc[g], 1.0f);
#pragma unroll 8
    for (int c = 0; c < 64; ++c)
      atomicAdd(&ls[g * 64 + c], h2[(size_t)node * 64 + c]);
  }
  __syncthreads();
  for (int i = threadIdx.x; i < NGRAPHS * 64; i += blockDim.x)
    atomicAdd(&pool[i], ls[i]);
  if (threadIdx.x < NGRAPHS) atomicAdd(&cnt[threadIdx.x], lc[threadIdx.x]);
}

// ---------------- final classifier: out[8,10] = (pool/cnt) @ W + b ----------
__global__ void k_final(const float* __restrict__ pool, const float* __restrict__ cnt,
                        const float* __restrict__ W, const float* __restrict__ b,
                        float* __restrict__ out) {
  int i = threadIdx.x;
  if (i >= NGRAPHS * NCLASSES) return;
  int g = i / NCLASSES, k = i - g * NCLASSES;
  float inv = 1.f / fmaxf(cnt[g], 1.0f);
  float acc = b[k];
#pragma unroll
  for (int c = 0; c < 64; ++c)
    acc += (pool[g * 64 + c] * inv) * W[c * 10 + k];
  out[i] = acc;
}

extern "C" void kernel_launch(void* const* d_in, const int* in_sizes, int n_in,
                              void* d_out, int out_size, void* d_ws, size_t ws_size,
                              hipStream_t stream) {
  const float* x    = (const float*)d_in[0];
  const int*   ei   = (const int*)d_in[1];
  const int*   bat  = (const int*)d_in[2];
  const float* fc0w = (const float*)d_in[3];
  const float* fc0b = (const float*)d_in[4];
  const float* u1   = (const float*)d_in[5];
  const float* c1   = (const float*)d_in[6];
  const float* w1   = (const float*)d_in[7];
  const float* b1   = (const float*)d_in[8];
  const float* u2   = (const float*)d_in[9];
  const float* c2   = (const float*)d_in[10];
  const float* w2   = (const float*)d_in[11];
  const float* b2   = (const float*)d_in[12];
  const float* fc1w = (const float*)d_in[13];
  const float* fc1b = (const float*)d_in[14];
  const int* srcI = ei;            // edge_index[0] = src (x_j)
  const int* dstI = ei + NEDGES;   // edge_index[1] = dst (x_i)
  float* out = (float*)d_out;

  // workspace layout (floats); xw buffer is shared between the two layers
  float* ws = (float*)d_ws;
  size_t off = 0;
  float* h0   = ws + off; off += (size_t)NNODES * 16;
  float* xw   = ws + off; off += (size_t)NNODES * 256;
  float* q    = ws + off; off += (size_t)NEDGES * 4;
  float* h1   = ws + off; off += (size_t)NNODES * 32;   // agg1, then h1 in place
  float* h2   = ws + off; off += (size_t)NNODES * 64;   // agg2, then h2 in place
  float* deg  = ws + off; off += (size_t)NNODES;
  float* pool = ws + off; off += NGRAPHS * 64;
  float* cnt  = ws + off; off += NGRAPHS;
  (void)ws_size; (void)in_sizes; (void)n_in; (void)out_size;

  // zero accumulation buffers (graph-capture safe)
  hipMemsetAsync(h1,   0, (size_t)NNODES * 32 * sizeof(float), stream);
  hipMemsetAsync(h2,   0, (size_t)NNODES * 64 * sizeof(float), stream);
  hipMemsetAsync(deg,  0, (size_t)NNODES * sizeof(float), stream);
  hipMemsetAsync(pool, 0, (NGRAPHS * 64 + NGRAPHS) * sizeof(float), stream);

  const int B = 256;
  // fc0 + degree
  k_fc0<<<(NNODES + B - 1) / B, B, 0, stream>>>(x, fc0w, fc0b, h0, NNODES);
  k_deg<<<(NEDGES + B - 1) / B, B, 0, stream>>>(dstI, deg, NEDGES);

  int gemm_blocks = ((NNODES / 16) + 7) / 8;  // 8 waves per 256-thread block

  // ---- layer 1: heads=4, out_c=32 (OUT total = 128), K = 16 ----
  k_gemm_wmma<16, 128><<<gemm_blocks, B, 0, stream>>>(h0, w1, xw, NNODES);
  k_q<16><<<(NEDGES + B - 1) / B, B, 0, stream>>>(h0, srcI, dstI, u1, c1, q, NEDGES);
  k_edge<32><<<(NEDGES * 32 + B - 1) / B, B, 0, stream>>>(q, xw, srcI, dstI, h1, NEDGES);
  k_norm<32><<<(NNODES * 32 + B - 1) / B, B, 0, stream>>>(h1, deg, b1, NNODES);

  // ---- layer 2: heads=4, out_c=64 (OUT total = 256), K = 32 ----
  k_gemm_wmma<32, 256><<<gemm_blocks, B, 0, stream>>>(h1, w2, xw, NNODES);
  k_q<32><<<(NEDGES + B - 1) / B, B, 0, stream>>>(h1, srcI, dstI, u2, c2, q, NEDGES);
  k_edge<64><<<(NEDGES * 32 + B - 1) / B, B, 0, stream>>>(q, xw, srcI, dstI, h2, NEDGES);
  k_norm<64><<<(NNODES * 64 + B - 1) / B, B, 0, stream>>>(h2, deg, b2, NNODES);

  // ---- pool + classifier ----
  k_pool<<<(NNODES + B - 1) / B, B, 0, stream>>>(h2, bat, pool, cnt, NNODES);
  k_final<<<1, 128, 0, stream>>>(pool, cnt, fc1w, fc1b, out);
}
